// MultiHead_Attention_17411797418205
// MI455X (gfx1250) — matrix operand order repaired and checked by reading, not compiled, not run on hardware
//
#include <hip/hip_runtime.h>
#include <hip/hip_bf16.h>
#include <stdint.h>

#define D_MODEL   1024
#define NUM_HEADS 16
#define DK        64
#define SEQ       2048
#define BATCH     2

typedef __bf16 bf16_t;
typedef __attribute__((ext_vector_type(16))) __bf16 v16bf;
typedef __attribute__((ext_vector_type(8)))  float  v8f;
typedef unsigned int u32x4 __attribute__((ext_vector_type(4)));
typedef int          i32x8 __attribute__((ext_vector_type(8)));
typedef int          i32x4 __attribute__((ext_vector_type(4)));

// ---------- helpers ----------

__device__ __forceinline__ bf16_t f2bf(float x) {
  union { float f; uint32_t u; } v; v.f = x;
  uint32_t r = v.u + 0x7FFFu + ((v.u >> 16) & 1u);   // round-to-nearest-even
  union { uint16_t u; bf16_t b; } o; o.u = (uint16_t)(r >> 16);
  return o.b;
}

__device__ __forceinline__ v8f vzero8() {
  v8f v;
#pragma unroll
  for (int i = 0; i < 8; ++i) v[i] = 0.0f;
  return v;
}

// A fragment (16x32 bf16, MxK): lane holds row M=lane%16; element e maps to
// K = kbase + 16*(e/8) + 8*g + (e%8)  -> two contiguous 16B loads.
__device__ __forceinline__ v16bf load_frag_a(const bf16_t* row, int kbase, int g) {
  union { v16bf v; uint4 q[2]; } u;
  u.q[0] = *(const uint4*)(row + kbase + 8 * g);
  u.q[1] = *(const uint4*)(row + kbase + 16 + 8 * g);
  return u.v;
}

// B fragment (32x16 bf16, KxN): lane holds col N=lane%16; element e maps to
// K = kbase + 16*(e/8) + 8*g + (e%8)  -> two 16B loads (B stored column-major in LDS).
__device__ __forceinline__ v16bf load_frag_b(const bf16_t* row, int kbase, int g) {
  union { v16bf v; uint4 q[2]; } u;
  const bf16_t* p = row + kbase + 8 * g;
  u.q[0] = *(const uint4*)(p);
  u.q[1] = *(const uint4*)(p + 16);
  return u.v;
}

__device__ __forceinline__ v8f wmma_bf16(v16bf a, v16bf b, v8f c) {
  // (neg_a, A, neg_b, B, c_mod, C, reuse_a, reuse_b)
  return __builtin_amdgcn_wmma_f32_16x16x32_bf16(false, a, false, b, (short)0, c,
                                                 false, false);
}

// ---------- Tensor Data Mover: 2D bf16 tile, global -> LDS ----------
// D# per CDNA5 ISA. data_size=1 (2 bytes/elem).
// Tile = [tile_d1 rows][tile_d0 cols], row stride = stride0 elements.
// LDS receives tile rows packed contiguously ([tile_d1][tile_d0]).
// 6-arg builtin form (amdgpu-toolchain / clang-23).
__device__ __forceinline__ void tdm_load_2d_bf16(
    unsigned lds_addr, const void* gaddr,
    unsigned tensor_d0, unsigned tensor_d1,
    unsigned tile_d0, unsigned tile_d1, unsigned stride0)
{
  uint64_t ga = (uint64_t)(uintptr_t)gaddr;
  u32x4 g0;
  g0[0] = 1u;                                        // count=1 (valid user D#)
  g0[1] = lds_addr;                                  // lds_addr [63:32]
  g0[2] = (unsigned)(ga & 0xFFFFFFFFu);              // global_addr[31:0]
  g0[3] = (unsigned)((ga >> 32) & 0x01FFFFFFu)       // global_addr[56:32]
          | (2u << 30);                              // type=2 ("image") [127:126]
  i32x8 g1;
  g1[0] = (int)(1u << 16);                           // data_size=1 (2B); wg_mask=0
  g1[1] = (int)((tensor_d0 & 0xFFFFu) << 16);        // tensor_dim0[15:0] @ [63:48]
  g1[2] = (int)(((tensor_d0 >> 16) & 0xFFFFu)        // tensor_dim0[31:16]
          | ((tensor_d1 & 0xFFFFu) << 16));          // tensor_dim1[15:0]
  g1[3] = (int)(((tensor_d1 >> 16) & 0xFFFFu)        // tensor_dim1[31:16]
          | ((tile_d0 & 0xFFFFu) << 16));            // tile_dim0 @ [127:112]
  g1[4] = (int)(tile_d1 & 0xFFFFu);                  // tile_dim1; tile_dim2=0
  g1[5] = (int)stride0;                              // tensor_dim0_stride[31:0]
  g1[6] = 0;                                         // stride0[47:32], dim1_stride lo
  g1[7] = 0;
  i32x4 gz4  = {0, 0, 0, 0};                         // 2D: groups 2/3 unused
  i32x8 gz8  = {0, 0, 0, 0, 0, 0, 0, 0};
  __builtin_amdgcn_tensor_load_to_lds(g0, g1, gz4, gz4, gz8, 0);
}

// ---------- GEMM: C[M,N] = A[M,K] * W[K,N] + bias ----------
// 128x128 tile per block, 256 threads (8 waves, 4x2 wave grid, 32x64 per wave).
template <bool A_BF16, bool HEAD_SPLIT>
__global__ __launch_bounds__(256) void gemm_wmma(
    const void* __restrict__ Aptr, const float* __restrict__ W,
    const float* __restrict__ bias, void* __restrict__ Out,
    int M, int N, int K)
{
  constexpr int TM = 128, TN = 128, TK = 32, PAD = 8;
  __shared__ bf16_t lA[TM][TK + PAD];   // [m][k]
  __shared__ bf16_t lB[TN][TK + PAD];   // W tile transposed: [n][k]

  const int tid  = threadIdx.x;
  const int lane = tid & 31;
  const int w    = tid >> 5;
  const int wm   = w >> 1;       // 0..3 (M direction)
  const int wn   = w & 1;        // 0..1 (N direction)
  const int g    = lane >> 4;
  const int ln   = lane & 15;
  const int mBase = blockIdx.y * TM;
  const int nBase = blockIdx.x * TN;

  v8f acc[2][4];
#pragma unroll
  for (int i = 0; i < 2; ++i)
#pragma unroll
    for (int j = 0; j < 4; ++j) acc[i][j] = vzero8();

  for (int k0 = 0; k0 < K; k0 += TK) {
    __syncthreads();
    if (A_BF16) {
      const bf16_t* A = (const bf16_t*)Aptr;
#pragma unroll
      for (int i = 0; i < 2; ++i) {
        int linear = tid + i * 256;          // 512 chunks of 8 halves
        int row = linear >> 2, c8 = linear & 3;
        *(uint4*)(&lA[row][c8 * 8]) =
            *(const uint4*)(A + (size_t)(mBase + row) * K + k0 + c8 * 8);
      }
    } else {
      const float* A = (const float*)Aptr;
#pragma unroll
      for (int i = 0; i < 4; ++i) {
        int linear = tid + i * 256;          // 1024 float4 chunks
        int row = linear >> 3, c4 = linear & 7;
        float4 d = *(const float4*)(A + (size_t)(mBase + row) * K + k0 + c4 * 4);
        bf16_t* dst = &lA[row][c4 * 4];
        dst[0] = f2bf(d.x); dst[1] = f2bf(d.y);
        dst[2] = f2bf(d.z); dst[3] = f2bf(d.w);
      }
    }
    // W tile 32x128 fp32, stored transposed (column-major for WMMA B).
#pragma unroll
    for (int i = 0; i < 4; ++i) {
      int linear = tid + i * 256;            // 1024 float4 chunks
      int kr = linear >> 5, c4 = linear & 31;
      float4 d = *(const float4*)(W + (size_t)(k0 + kr) * N + nBase + c4 * 4);
      lB[c4 * 4 + 0][kr] = f2bf(d.x);
      lB[c4 * 4 + 1][kr] = f2bf(d.y);
      lB[c4 * 4 + 2][kr] = f2bf(d.z);
      lB[c4 * 4 + 3][kr] = f2bf(d.w);
    }
    __syncthreads();

    if (k0 + TK < K) {  // gfx1250 global_prefetch_b8 hints for next tiles
      const char* pfA = A_BF16
        ? (const char*)((const bf16_t*)Aptr + (size_t)(mBase + (tid >> 1)) * K + k0 + TK)
        : (const char*)((const float*)Aptr + (size_t)(mBase + (tid >> 1)) * K + k0 + TK);
      __builtin_prefetch(pfA, 0, 3);
      const char* pfW =
          (const char*)(W + (size_t)(k0 + TK + (tid >> 3)) * N + nBase + (tid & 7) * 16);
      __builtin_prefetch(pfW, 0, 3);
    }

    v16bf afr[2], bfr[4];
#pragma unroll
    for (int mf = 0; mf < 2; ++mf)
      afr[mf] = load_frag_a(&lA[wm * 32 + mf * 16 + ln][0], 0, g);
#pragma unroll
    for (int nf = 0; nf < 4; ++nf)
      bfr[nf] = load_frag_b(&lB[wn * 64 + nf * 16 + ln][0], 0, g);
#pragma unroll
    for (int mf = 0; mf < 2; ++mf)
#pragma unroll
      for (int nf = 0; nf < 4; ++nf)
        acc[mf][nf] = wmma_bf16(afr[mf], bfr[nf], acc[mf][nf]);
  }

  // epilogue: C-fragment row = r + 8*g, col = lane%16
#pragma unroll
  for (int mf = 0; mf < 2; ++mf) {
#pragma unroll
    for (int nf = 0; nf < 4; ++nf) {
      const int n = nBase + wn * 64 + nf * 16 + ln;
      const float bv = bias[n];
#pragma unroll
      for (int r = 0; r < 8; ++r) {
        const int m = mBase + wm * 32 + mf * 16 + r + 8 * g;
        const float val = acc[mf][nf][r] + bv;
        if (HEAD_SPLIT) {
          bf16_t* O = (bf16_t*)Out;                 // [B,H,S,Dk] bf16
          const int bb = m / SEQ, s = m % SEQ;
          const int h = n / DK, dk = n % DK;
          O[(((size_t)bb * NUM_HEADS + h) * SEQ + s) * DK + dk] = f2bf(val);
        } else {
          ((float*)Out)[(size_t)m * N + n] = val;   // [M,N] fp32
        }
      }
    }
  }
}

// ---------- Flash attention over bf16 heads ----------
// Block = 256 threads (8 waves). Block handles one (b,h) and 128 query rows;
// wave w owns query rows [w*16, w*16+16). Key tiles of 128, online softmax.
// Q and K tiles come in via the Tensor Data Mover (TENSORcnt), issued by wave 0.
__global__ __launch_bounds__(256) void flash_attn(
    const bf16_t* __restrict__ Qh, const bf16_t* __restrict__ Kh,
    const bf16_t* __restrict__ Vh, bf16_t* __restrict__ ctxOut)
{
  __shared__ bf16_t lK[128][DK];        // [key][d]   16 KB (TDM destination)
  __shared__ bf16_t lVT[DK][128];       // [d][key]   16 KB (V transposed)
  __shared__ bf16_t lP[8][16][128];     // per-wave P staging, 32 KB (Q staging at start)

  const int tid  = threadIdx.x;
  const int lane = tid & 31;
  const int w    = tid >> 5;
  const int g    = lane >> 4;
  const int ln   = lane & 15;
  const int qBase = blockIdx.x * 128;
  const int bh    = blockIdx.y;          // b*H + h
  const size_t headOff = (size_t)bh * SEQ * DK;
  const bf16_t* Q  = Qh + headOff;
  const bf16_t* Kp = Kh + headOff;
  const bf16_t* Vp = Vh + headOff;

  // ---- stage Q tile [128][64] in lP area via TDM, pull per-wave A fragments
  bf16_t* lQ = &lP[0][0][0];
  if (w == 0) {
    tdm_load_2d_bf16((unsigned)(uintptr_t)lQ, Q + (size_t)qBase * DK,
                     DK, SEQ, DK, 128, DK);
    __builtin_amdgcn_s_wait_tensorcnt(0);
  }
  __syncthreads();
  v16bf qf[2];
  {
    const bf16_t* qrow = lQ + (w * 16 + ln) * DK;
    qf[0] = load_frag_a(qrow, 0, g);
    qf[1] = load_frag_a(qrow, 32, g);
  }
  __syncthreads();                        // lP now free for P staging

  float m_i[8], l_i[8];
#pragma unroll
  for (int r = 0; r < 8; ++r) { m_i[r] = -3.0e38f; l_i[r] = 0.0f; }
  v8f ctx[4];
#pragma unroll
  for (int d = 0; d < 4; ++d) ctx[d] = vzero8();

  for (int kt = 0; kt < SEQ; kt += 128) {
    __syncthreads();                      // all readers of lK/lVT done
    // K tile via TDM (wave 0 only; TDM ignores EXEC, tracked on TENSORcnt)
    if (w == 0)
      tdm_load_2d_bf16((unsigned)(uintptr_t)&lK[0][0], Kp + (size_t)kt * DK,
                       DK, SEQ, DK, 128, DK);
    // V tile transposed into lVT[d][key] by all waves (overlaps the TDM)
#pragma unroll
    for (int i = 0; i < 4; ++i) {
      int linear = tid + i * 256;
      int row = linear >> 3, c8 = linear & 7;
      uint4 dv = *(const uint4*)(Vp + (size_t)(kt + row) * DK + c8 * 8);
      const bf16_t* pv = (const bf16_t*)&dv;
#pragma unroll
      for (int j = 0; j < 8; ++j) lVT[c8 * 8 + j][row] = pv[j];
    }
    if (w == 0) __builtin_amdgcn_s_wait_tensorcnt(0);
    __syncthreads();

    // scores S = Q * K^T  (8 fragments of 16 keys each, K-dim = 64 = 2 steps)
    v8f sfr[8];
#pragma unroll
    for (int nf = 0; nf < 8; ++nf) sfr[nf] = vzero8();
#pragma unroll
    for (int kk = 0; kk < 2; ++kk) {
#pragma unroll
      for (int nf = 0; nf < 8; ++nf) {
        v16bf kb = load_frag_b(&lK[nf * 16 + ln][0], kk * 32, g);
        sfr[nf] = wmma_bf16(qf[kk], kb, sfr[nf]);
      }
    }

    // online softmax (row M = r + 8g lives in the 16-lane half 'g')
    const float scale = 0.125f;           // 1/sqrt(Dk)
    float rmax[8], rsum[8], alpha[8];
#pragma unroll
    for (int r = 0; r < 8; ++r) rmax[r] = -3.0e38f;
#pragma unroll
    for (int nf = 0; nf < 8; ++nf)
#pragma unroll
      for (int r = 0; r < 8; ++r) {
        sfr[nf][r] *= scale;
        rmax[r] = fmaxf(rmax[r], sfr[nf][r]);
      }
#pragma unroll
    for (int r = 0; r < 8; ++r)
#pragma unroll
      for (int off = 1; off < 16; off <<= 1)
        rmax[r] = fmaxf(rmax[r], __shfl_xor(rmax[r], off, 32));
#pragma unroll
    for (int r = 0; r < 8; ++r) {
      float mn = fmaxf(m_i[r], rmax[r]);
      alpha[r] = __expf(m_i[r] - mn);
      m_i[r] = mn;
      rsum[r] = 0.0f;
    }
#pragma unroll
    for (int nf = 0; nf < 8; ++nf)
#pragma unroll
      for (int r = 0; r < 8; ++r) {
        float p = __expf(sfr[nf][r] - m_i[r]);
        sfr[nf][r] = p;
        rsum[r] += p;
      }
#pragma unroll
    for (int r = 0; r < 8; ++r)
#pragma unroll
      for (int off = 1; off < 16; off <<= 1)
        rsum[r] += __shfl_xor(rsum[r], off, 32);
#pragma unroll
    for (int r = 0; r < 8; ++r) l_i[r] = l_i[r] * alpha[r] + rsum[r];
#pragma unroll
    for (int d = 0; d < 4; ++d)
#pragma unroll
      for (int r = 0; r < 8; ++r) ctx[d][r] *= alpha[r];

    // restage P (C layout) -> per-wave LDS -> A layout fragments
#pragma unroll
    for (int nf = 0; nf < 8; ++nf)
#pragma unroll
      for (int r = 0; r < 8; ++r)
        lP[w][r + 8 * g][nf * 16 + ln] = f2bf(sfr[nf][r]);

    // ctx += P * V   (K-dim = 128 keys = 4 WMMA steps)
#pragma unroll
    for (int kk2 = 0; kk2 < 4; ++kk2) {
      v16bf pa = load_frag_a(&lP[w][ln][0], kk2 * 32, g);
#pragma unroll
      for (int d = 0; d < 4; ++d) {
        v16bf vb = load_frag_b(&lVT[d * 16 + ln][0], kk2 * 32, g);
        ctx[d] = wmma_bf16(pa, vb, ctx[d]);
      }
    }
  }

  // epilogue: normalize and store ctx as [B,S,H*Dk] bf16 (ready for out-proj GEMM)
  const int b = bh / NUM_HEADS, h = bh % NUM_HEADS;
#pragma unroll
  for (int d = 0; d < 4; ++d)
#pragma unroll
    for (int r = 0; r < 8; ++r) {
      const int qrow = qBase + w * 16 + r + 8 * g;
      const int dk = d * 16 + ln;
      const float v = ctx[d][r] / l_i[r];
      ctxOut[((size_t)b * SEQ + qrow) * D_MODEL + h * DK + dk] = f2bf(v);
    }
}

// ---------- launch ----------

extern "C" void kernel_launch(void* const* d_in, const int* in_sizes, int n_in,
                              void* d_out, int out_size, void* d_ws, size_t ws_size,
                              hipStream_t stream) {
  const float* q   = (const float*)d_in[0];
  const float* k   = (const float*)d_in[1];
  const float* v   = (const float*)d_in[2];
  const float* W_q = (const float*)d_in[3];
  const float* b_q = (const float*)d_in[4];
  const float* W_k = (const float*)d_in[5];
  const float* b_k = (const float*)d_in[6];
  const float* W_v = (const float*)d_in[7];
  const float* b_v = (const float*)d_in[8];
  const float* W_o = (const float*)d_in[9];
  const float* b_o = (const float*)d_in[10];
  float* out = (float*)d_out;

  const size_t elems = (size_t)BATCH * SEQ * D_MODEL;   // 4M elements
  bf16_t* Qh  = (bf16_t*)d_ws;
  bf16_t* Kh  = Qh + elems;
  bf16_t* Vh  = Kh + elems;
  bf16_t* Ctx = Vh + elems;                              // 32 MB total bf16

  const int M = BATCH * SEQ;           // 4096
  dim3 gridP(D_MODEL / 128, M / 128);  // (8, 32)
  dim3 blk(256);

  gemm_wmma<false, true><<<gridP, blk, 0, stream>>>(q, W_q, b_q, Qh, M, D_MODEL, D_MODEL);
  gemm_wmma<false, true><<<gridP, blk, 0, stream>>>(k, W_k, b_k, Kh, M, D_MODEL, D_MODEL);
  gemm_wmma<false, true><<<gridP, blk, 0, stream>>>(v, W_v, b_v, Vh, M, D_MODEL, D_MODEL);

  dim3 gridA(SEQ / 128, BATCH * NUM_HEADS);              // (16, 32)
  flash_attn<<<gridA, blk, 0, stream>>>(Qh, Kh, Vh, Ctx);

  gemm_wmma<true, false><<<gridP, blk, 0, stream>>>(Ctx, W_o, b_o, out, M, D_MODEL, D_MODEL);
}
